// FlexSentenceEncoder_25881472926382
// MI455X (gfx1250) — matrix operand order repaired and checked by reading, not compiled, not run on hardware
//
#include <hip/hip_runtime.h>
#include <math.h>

typedef _Float16 v16h __attribute__((ext_vector_type(16)));
typedef _Float16 v8h  __attribute__((ext_vector_type(8)));
typedef _Float16 v2h  __attribute__((ext_vector_type(2)));
typedef float    v8f  __attribute__((ext_vector_type(8)));

#define PAD_IDX 1
#define SLEN 2048
#define DIM  768
#define NH   12
#define HD   64
#define FFD  3072
#define NL   6
#define WINR 128

// ---------------- positions: fairseq make_positions (serial, one-shot) -------
__global__ void positions_kernel(const int* __restrict__ tokens, int* __restrict__ pos) {
  if (threadIdx.x == 0 && blockIdx.x == 0) {
    int cum = 0;
    for (int s = 0; s < SLEN; ++s) {
      int np = (tokens[s] != PAD_IDX) ? 1 : 0;
      cum += np;
      pos[s] = (PAD_IDX + cum) * np;
    }
  }
}

// ---------------- embedding sum + pad zeroing --------------------------------
__global__ void embed_kernel(const int* __restrict__ tokens, const int* __restrict__ seg,
                             const int* __restrict__ pos,
                             const float* __restrict__ et, const float* __restrict__ pe,
                             const float* __restrict__ se, float* __restrict__ x) {
  const int s = blockIdx.x;
  const int t = tokens[s];
  const float nonpad = (t != PAD_IDX) ? 1.0f : 0.0f;
  const int p = pos[s];
  const int g = seg[s];
  for (int d = threadIdx.x; d < DIM; d += blockDim.x) {
    float v = et[(size_t)t * DIM + d] + pe[(size_t)p * DIM + d] + se[(size_t)g * DIM + d];
    x[(size_t)s * DIM + d] = v * nonpad;
  }
}

// ---------------- WMMA GEMM: Out = act((A[MxK] * W[KxN] + bias) * alpha) -----
// Block = 256 threads (8 waves). Block tile 64(M) x 128(N), K step 32.
// Waves arranged 4(M) x 2(N); each wave owns a 16x64 strip = 4 WMMA tiles,
// reusing one A fragment across 4 B fragments (4 v_wmma per K-step per wave).
// LDS holds f16 tiles: A row-major (Ah[m][k]), B transposed (Bh[n][k]) so every
// fragment is contiguous 16B runs -> ds_load_b128.
#define BM 64
#define BN 128
#define BK 32
#define LDP 40   // row pitch in halves (80 B, multiple of 16 B)

__global__ __launch_bounds__(256) void gemm_wmma(
    const float* __restrict__ A, const float* __restrict__ W,
    const float* __restrict__ bias, float* __restrict__ Out,
    int M, int N, int K, float alpha, int relu)
{
  __shared__ _Float16 Ah[BM][LDP];   // 64 x 40 halves
  __shared__ _Float16 Bh[BN][LDP];   // 128 x 40 halves

  const int tid  = threadIdx.x;
  const int lane = tid & 31;
  const int wave = tid >> 5;
  const int wm   = wave >> 1;        // 0..3
  const int wn   = wave & 1;         // 0..1
  const int m0   = blockIdx.x * BM;
  const int n0   = blockIdx.y * BN;

  v8f acc[4] = {};

  const int mrow  = wm * 16 + (lane & 15);
  const int kbase = (lane < 16) ? 0 : 8;    // A: K half-run start
  const int kb0   = (lane < 16) ? 0 : 8;    // B: K half-run start

  // A staging map: 1024 float2 pairs; thread t handles pairs t, t+256, ...
  // B staging map: col c = tid&127, start row tid>>7, step 2.
  const int bc = tid & 127;
  const int br = tid >> 7;

  for (int k0 = 0; k0 < K; k0 += BK) {
    // prefetch next K tiles into cache hierarchy (global_prefetch_b8)
    if (k0 + BK < K) {
      __builtin_prefetch(&A[(size_t)(m0 + (tid >> 2)) * K + (k0 + BK)], 0, 3);
      __builtin_prefetch(&W[(size_t)(k0 + BK + (tid >> 5)) * N + n0 + (tid & 31) * 4], 0, 3);
    }

    // stage A 64x32 fp32 -> f16 (vectorized: float2 -> packed half2)
    #pragma unroll
    for (int i = 0; i < 4; ++i) {
      int p  = tid + i * 256;       // 0..1023
      int r  = p >> 4;              // 0..63
      int kp = (p & 15) * 2;        // 0,2,..,30
      const float2 v = *reinterpret_cast<const float2*>(
          &A[(size_t)(m0 + r) * K + (k0 + kp)]);
      v2h h; h[0] = (_Float16)v.x; h[1] = (_Float16)v.y;
      *reinterpret_cast<__shared__ v2h*>(&Ah[r][kp]) = h;
    }
    // stage B 32x128 fp32 -> f16 transposed into Bh[n][k]
    #pragma unroll
    for (int r = br; r < BK; r += 2) {
      float v = W[(size_t)(k0 + r) * N + (n0 + bc)];
      Bh[bc][r] = (_Float16)v;
    }
    __syncthreads();

    // A fragment: lanes 0-15 -> M=lane, K {kbase..kbase+7, kbase+16..kbase+23}
    v8h alo = *reinterpret_cast<const __shared__ v8h*>(&Ah[mrow][kbase]);
    v8h ahi = *reinterpret_cast<const __shared__ v8h*>(&Ah[mrow][kbase + 16]);
    v16h af = __builtin_shufflevector(alo, ahi,
        0, 1, 2, 3, 4, 5, 6, 7, 8, 9, 10, 11, 12, 13, 14, 15);

    #pragma unroll
    for (int jn = 0; jn < 4; ++jn) {
      const int nn = wn * 64 + jn * 16 + (lane & 15);
      v8h blo = *reinterpret_cast<const __shared__ v8h*>(&Bh[nn][kb0]);
      v8h bhi = *reinterpret_cast<const __shared__ v8h*>(&Bh[nn][kb0 + 16]);
      v16h bf = __builtin_shufflevector(blo, bhi,
          0, 1, 2, 3, 4, 5, 6, 7, 8, 9, 10, 11, 12, 13, 14, 15);
      acc[jn] = __builtin_amdgcn_wmma_f32_16x16x32_f16(
          false, af, false, bf, (short)0, acc[jn], false, false);
    }
    __syncthreads();
  }

  // D layout: VGPR r -> M = r + 8*(lane>=16); N = lane&15
  const int mhi = (lane < 16) ? 0 : 8;
  #pragma unroll
  for (int jn = 0; jn < 4; ++jn) {
    const int nc = n0 + wn * 64 + jn * 16 + (lane & 15);
    const float bv = bias[nc];
    #pragma unroll
    for (int r = 0; r < 8; ++r) {
      int mr = m0 + wm * 16 + r + mhi;
      float v = (acc[jn][r] + bv) * alpha;
      if (relu) v = fmaxf(v, 0.0f);
      Out[(size_t)mr * N + nc] = v;
    }
  }
}

// ---------------- relative position bucket (bidirectional, NB=32, MAXD=128) --
__device__ __forceinline__ int rel_bucket(int rp) {
  int n = -rp;
  int ret = (n < 0) ? 16 : 0;
  n = (n < 0) ? -n : n;
  if (n < 8) return ret + n;
  int v = 8 + (int)(__logf((float)n * 0.125f) * (1.0f / __logf(16.0f)) * 8.0f);
  if (v > 15) v = 15;
  return ret + v;
}

// ---------------- windowed attention, one wave per (q, h) --------------------
__global__ void attn_kernel(const float* __restrict__ Q, const float* __restrict__ Km,
                            const float* __restrict__ Vm, const int* __restrict__ tokens,
                            const float* __restrict__ rel_bias, float* __restrict__ Outp)
{
  const int q    = blockIdx.x;
  const int h    = blockIdx.y;
  const int lane = threadIdx.x;          // 0..31, two dims per lane
  const int base = h * HD;

  const float q0 = Q[(size_t)q * DIM + base + lane];
  const float q1 = Q[(size_t)q * DIM + base + lane + 32];

  float m = -3.0e38f, l = 0.0f, a0 = 0.0f, a1 = 0.0f;
  int lo = q - WINR; if (lo < 0) lo = 0;
  int hi = q + WINR; if (hi > SLEN - 1) hi = SLEN - 1;

  for (int k = lo; k <= hi; ++k) {
    float p = q0 * Km[(size_t)k * DIM + base + lane]
            + q1 * Km[(size_t)k * DIM + base + lane + 32];
    #pragma unroll
    for (int off = 16; off > 0; off >>= 1) p += __shfl_xor(p, off, 32);

    int bkt = rel_bucket(k - q);
    float s = p + rel_bias[bkt * NH + h];
    if (tokens[k] == PAD_IDX) s -= 1.0e30f;

    float mn = fmaxf(m, s);
    float c  = __expf(m - mn);
    float w  = __expf(s - mn);
    a0 = a0 * c + w * Vm[(size_t)k * DIM + base + lane];
    a1 = a1 * c + w * Vm[(size_t)k * DIM + base + lane + 32];
    l  = l * c + w;
    m  = mn;
  }
  Outp[(size_t)q * DIM + base + lane]      = a0 / l;
  Outp[(size_t)q * DIM + base + lane + 32] = a1 / l;
}

// ---------------- fused residual add + LayerNorm, block per row --------------
__global__ __launch_bounds__(256) void ln_kernel(
    const float* __restrict__ resid, const float* __restrict__ delta,
    const float* __restrict__ gamma, const float* __restrict__ beta,
    float* __restrict__ out)
{
  __shared__ float red[256];
  __shared__ float red2[256];
  const int s = blockIdx.x;
  float local[3];
  float sum = 0.0f, sq = 0.0f;
  #pragma unroll
  for (int i = 0; i < 3; ++i) {
    int d = threadIdx.x + i * 256;
    float v = resid[(size_t)s * DIM + d] + delta[(size_t)s * DIM + d];
    local[i] = v; sum += v; sq += v * v;
  }
  red[threadIdx.x] = sum; red2[threadIdx.x] = sq;
  __syncthreads();
  for (int off = 128; off > 0; off >>= 1) {
    if (threadIdx.x < off) {
      red[threadIdx.x]  += red[threadIdx.x + off];
      red2[threadIdx.x] += red2[threadIdx.x + off];
    }
    __syncthreads();
  }
  const float mean = red[0] * (1.0f / DIM);
  const float var  = red2[0] * (1.0f / DIM) - mean * mean;
  const float inv  = rsqrtf(var + 1e-5f);
  #pragma unroll
  for (int i = 0; i < 3; ++i) {
    int d = threadIdx.x + i * 256;
    out[(size_t)s * DIM + d] = (local[i] - mean) * inv * gamma[d] + beta[d];
  }
}

// ---------------- host orchestration ----------------------------------------
extern "C" void kernel_launch(void* const* d_in, const int* in_sizes, int n_in,
                              void* d_out, int out_size, void* d_ws, size_t ws_size,
                              hipStream_t stream) {
  (void)in_sizes; (void)n_in; (void)out_size; (void)ws_size;

  const int*   tokens = (const int*)d_in[0];
  const int*   seg    = (const int*)d_in[1];
  const float* et     = (const float*)d_in[2];
  const float* pe     = (const float*)d_in[3];
  const float* se     = (const float*)d_in[4];
  const float* rb     = (const float*)d_in[5];
  const float* Wq = (const float*)d_in[6];   const float* bq  = (const float*)d_in[7];
  const float* Wk = (const float*)d_in[8];   const float* bk  = (const float*)d_in[9];
  const float* Wv = (const float*)d_in[10];  const float* bv  = (const float*)d_in[11];
  const float* Wo = (const float*)d_in[12];  const float* bo  = (const float*)d_in[13];
  const float* s1 = (const float*)d_in[14];  const float* sb1 = (const float*)d_in[15];
  const float* W1 = (const float*)d_in[16];  const float* fb1 = (const float*)d_in[17];
  const float* W2 = (const float*)d_in[18];  const float* fb2 = (const float*)d_in[19];
  const float* s2 = (const float*)d_in[20];  const float* sb2 = (const float*)d_in[21];

  char* ws = (char*)d_ws;
  const size_t SD = (size_t)SLEN * DIM * sizeof(float);
  float* X0  = (float*)(ws + 0 * SD);
  float* X1  = (float*)(ws + 1 * SD);
  float* XN  = (float*)(ws + 2 * SD);
  float* Qb  = (float*)(ws + 3 * SD);
  float* Kb  = (float*)(ws + 4 * SD);
  float* Vb  = (float*)(ws + 5 * SD);
  float* AT  = (float*)(ws + 6 * SD);
  float* PR  = (float*)(ws + 7 * SD);
  float* FFb = (float*)(ws + 8 * SD);
  int*   pos = (int*)(ws + 8 * SD + (size_t)SLEN * FFD * sizeof(float));

  positions_kernel<<<1, 32, 0, stream>>>(tokens, pos);
  embed_kernel<<<SLEN, 256, 0, stream>>>(tokens, seg, pos, et, pe, se, X0);

  const float qscale = 0.125f;   // 1/sqrt(HD=64)
  const dim3 gD(SLEN / BM, DIM / BN);   // 32 x 6
  const dim3 gF(SLEN / BM, FFD / BN);   // 32 x 24

  float* xin  = X0;
  float* xout = X1;
  for (int l = 0; l < NL; ++l) {
    const float* wq = Wq + (size_t)l * DIM * DIM;
    const float* wk = Wk + (size_t)l * DIM * DIM;
    const float* wv = Wv + (size_t)l * DIM * DIM;
    const float* wo = Wo + (size_t)l * DIM * DIM;
    const float* w1 = W1 + (size_t)l * DIM * FFD;
    const float* w2 = W2 + (size_t)l * FFD * DIM;

    gemm_wmma<<<gD, 256, 0, stream>>>(xin, wq, bq + (size_t)l * DIM, Qb, SLEN, DIM, DIM, qscale, 0);
    gemm_wmma<<<gD, 256, 0, stream>>>(xin, wk, bk + (size_t)l * DIM, Kb, SLEN, DIM, DIM, 1.0f, 0);
    gemm_wmma<<<gD, 256, 0, stream>>>(xin, wv, bv + (size_t)l * DIM, Vb, SLEN, DIM, DIM, 1.0f, 0);

    attn_kernel<<<dim3(SLEN, NH), 32, 0, stream>>>(Qb, Kb, Vb, tokens, rb, AT);

    gemm_wmma<<<gD, 256, 0, stream>>>(AT, wo, bo + (size_t)l * DIM, PR, SLEN, DIM, DIM, 1.0f, 0);
    ln_kernel<<<SLEN, 256, 0, stream>>>(xin, PR, s1 + (size_t)l * DIM, sb1 + (size_t)l * DIM, XN);

    gemm_wmma<<<gF, 256, 0, stream>>>(XN, w1, fb1 + (size_t)l * FFD, FFb, SLEN, FFD, DIM, 1.0f, 1);
    gemm_wmma<<<gD, 256, 0, stream>>>(FFb, w2, fb2 + (size_t)l * DIM, PR, SLEN, DIM, FFD, 1.0f, 0);

    float* dst = (l == NL - 1) ? (float*)d_out : xout;
    ln_kernel<<<SLEN, 256, 0, stream>>>(XN, PR, s2 + (size_t)l * DIM, sb2 + (size_t)l * DIM, dst);

    float* t = xin; xin = xout; xout = t;
  }
}
